// BaselineTNRDStage_74912819577231
// MI455X (gfx1250) — matrix operand order repaired and checked by reading, not compiled, not run on hardware
//
#include <hip/hip_runtime.h>
#include <hip/hip_bf16.h>

typedef __attribute__((ext_vector_type(16))) _Float16 v16h;
typedef __attribute__((ext_vector_type(8)))  float    v8f;

constexpr int BATCH = 8;
constexpr int HIMG  = 180;
constexpr int WIMG  = 180;
constexpr int NFILT = 24;
constexpr int NBAS  = 31;
constexpr int TOTAL = BATCH * HIMG * WIMG;          // 259200
constexpr int TS    = 16;                            // output tile
constexpr int HP    = TS + 4;                        // 20: scaled_phi halo
constexpr int HU    = TS + 8;                        // 24: u halo for conv1-of-halo
constexpr int USTR  = HU + 1;                        // 25: LDS row stride; col 24 = zero slot
constexpr int NPIX  = HP * HP;                       // 400 halo pixels
constexpr int MT    = NPIX / 16;                     // 25 M-tiles
constexpr int NTASK = MT * 2;                        // 2 N-tiles (filters 0-15, 16-23+pad)
constexpr int TILESX = (WIMG + TS - 1) / TS;         // 12

__device__ __forceinline__ void async_wait0() {
#if __has_builtin(__builtin_amdgcn_s_wait_asynccnt)
  __builtin_amdgcn_s_wait_asynccnt(0);
#else
  asm volatile("s_wait_asynccnt 0" ::: "memory");
#endif
}

// issue one per-lane async global->LDS dword copy (GVS mode: sgpr base + vgpr byte offset)
__device__ __forceinline__ void async_g2l_b32(uint32_t lds_byte_addr,
                                              uint32_t gbyte_off,
                                              const float* __restrict__ base) {
  asm volatile("global_load_async_to_lds_b32 %0, %1, %2"
               :: "v"(lds_byte_addr), "v"(gbyte_off), "s"(base)
               : "memory");
}

// ---------------- Kernel 1: 3x3 avg pool (count_include_pad) + block sums ----
__global__ __launch_bounds__(256) void k_avgpool(const float* __restrict__ u,
                                                 float* __restrict__ usig,
                                                 float* __restrict__ partial) {
  __shared__ float red[256];
  int idx = blockIdx.x * 256 + threadIdx.x;
  float v = 0.f;
  if (idx < TOTAL) {
    int b   = idx / (HIMG * WIMG);
    int rem = idx % (HIMG * WIMG);
    int y = rem / WIMG, x = rem % WIMG;
    const float* ub = u + b * HIMG * WIMG;
    float s = 0.f;
    #pragma unroll
    for (int dy = -1; dy <= 1; ++dy)
      #pragma unroll
      for (int dx = -1; dx <= 1; ++dx) {
        int yy = y + dy, xx = x + dx;
        if (yy >= 0 && yy < HIMG && xx >= 0 && xx < WIMG) s += ub[yy * WIMG + xx];
      }
    v = s * (1.f / 9.f);
    usig[idx] = v;
  }
  red[threadIdx.x] = v;
  __syncthreads();
  for (int off = 128; off > 0; off >>= 1) {
    if (threadIdx.x < off) red[threadIdx.x] += red[threadIdx.x + off];
    __syncthreads();
  }
  if (threadIdx.x == 0) partial[blockIdx.x] = red[0];
}

// ---------------- Kernel 2: reduce partials -> 1/M ---------------------------
__global__ __launch_bounds__(256) void k_reduce(const float* __restrict__ partial,
                                                int n, float* __restrict__ invM) {
  __shared__ float red[256];
  float s = 0.f;
  for (int i = threadIdx.x; i < n; i += 256) s += partial[i];
  red[threadIdx.x] = s;
  __syncthreads();
  for (int off = 128; off > 0; off >>= 1) {
    if (threadIdx.x < off) red[threadIdx.x] += red[threadIdx.x + off];
    __syncthreads();
  }
  if (threadIdx.x == 0) {
    float M = red[0] / (float)TOTAL + 0.001f;
    invM[0] = 1.f / M;
  }
}

// ---------------- Kernel 3: fused conv1(WMMA) -> RBF -> scale -> conv2 -------
__global__ __launch_bounds__(256) void k_fused(const float* __restrict__ u,
                                               const float* __restrict__ f,
                                               const float* __restrict__ filt,
                                               const float* __restrict__ lamp,
                                               const float* __restrict__ mu,
                                               const float* __restrict__ wts,
                                               const float* __restrict__ usig,
                                               const float* __restrict__ invMp,
                                               float* __restrict__ out) {
  __shared__ float s_u[HU * USTR];        // 600 f  : u halo (zero padded; [24] = zero slot)
  __shared__ float s_filt[NFILT * 25];    // 600 f  : filters
  __shared__ float s_usig[NPIX];          // 400 f  : u_sigma over halo (raw)
  __shared__ float s_mu[NBAS];
  __shared__ float s_w[NBAS];             // weights * invM (phi is linear in w)
  __shared__ float s_phi[NFILT * NPIX];   // 9600 f : scaled_phi halo tile

  const int tid = threadIdx.x;
  int blk  = blockIdx.x;
  int b    = blk / (TILESX * TILESX);
  int trem = blk % (TILESX * TILESX);
  int oy = (trem / TILESX) * TS;
  int ox = (trem % TILESX) * TS;
  const float* ub  = u    + b * HIMG * WIMG;
  const float* usb = usig + b * HIMG * WIMG;
  const float invM = invMp[0];

  // ---- phase 0: pre-zero LDS tiles (so async loads only cover in-bounds cells)
  for (int i = tid; i < HU * USTR; i += 256) s_u[i] = 0.f;
  for (int i = tid; i < NPIX; i += 256) s_usig[i] = 0.f;
  if (tid < NBAS) { s_mu[tid] = mu[tid]; s_w[tid] = wts[tid] * invM; }
  __syncthreads();  // DS zero-stores complete before ASYNC writes may land

  // ---- phase 1: async global -> LDS staging (ASYNCcnt-tracked, no VGPR round-trip)
  for (int i = tid; i < NFILT * 25; i += 256) {
    async_g2l_b32((uint32_t)(uintptr_t)(s_filt + i), (uint32_t)(i * 4), filt);
  }
  for (int i = tid; i < HU * HU; i += 256) {
    int r = i / HU, c = i % HU;
    int gy = oy - 4 + r, gx = ox - 4 + c;
    if (gy >= 0 && gy < HIMG && gx >= 0 && gx < WIMG) {
      async_g2l_b32((uint32_t)(uintptr_t)(s_u + r * USTR + c),
                    (uint32_t)((gy * WIMG + gx) * 4), ub);
    }
  }
  for (int i = tid; i < NPIX; i += 256) {
    int hy = i / HP, hx = i % HP;
    int gy = oy - 2 + hy, gx = ox - 2 + hx;
    if (gy >= 0 && gy < HIMG && gx >= 0 && gx < WIMG) {
      async_g2l_b32((uint32_t)(uintptr_t)(s_usig + i),
                    (uint32_t)((gy * WIMG + gx) * 4), usb);
    }
  }
  async_wait0();
  __syncthreads();

  const int lane = tid & 31;
  const int wv   = tid >> 5;          // 8 waves
  const int lm   = lane & 15;
  const int hi   = lane >> 4;

  // B fragments (filters), built once per wave. 32x16 f16 B in contiguous K order (same as A):
  // lane%16 = column N, K = s + 16*(lane/16)
  v16h bf0, bf1;
  #pragma unroll
  for (int s2 = 0; s2 < 16; ++s2) {
    int k = s2 + 16 * hi;
    float v0 = (k < 25) ? s_filt[lm * 25 + k] : 0.f;                  // filters 0..15
    float v1 = (lm < 8 && k < 25) ? s_filt[(16 + lm) * 25 + k] : 0.f; // filters 16..23
    bf0[s2] = (_Float16)v0;
    bf1[s2] = (_Float16)v1;
  }

  // implicit GEMM: [400 halo pixels x 25 taps] . [25 x 24 filters]
  for (int t = wv; t < NTASK; t += 8) {
    int mt = t >> 1, nt = t & 1;
    int p  = mt * 16 + lm;             // halo pixel, row M of A
    int hy = p / HP, hx = p % HP;
    // A layout 16x32 f16: lane%16 = row M; K halves split by lane/16.
    // Padding slots (k>=25) read the guaranteed-zero word s_u[24] -> no exec masking.
    v16h a;
    #pragma unroll
    for (int s2 = 0; s2 < 16; ++s2) {
      int k = s2 + 16 * hi;
      int idx = (k < 25) ? ((hy + k / 5) * USTR + (hx + k % 5)) : 24;
      a[s2] = (_Float16)s_u[idx];
    }
    v8f c = {};
    c = __builtin_amdgcn_wmma_f32_16x16x32_f16(false, a, false, (nt ? bf1 : bf0),
                                               (short)0, c, false, false);
    int n = nt * 16 + lm;              // filter channel (column N of D)
    #pragma unroll
    for (int r = 0; r < 8; ++r) {
      int pr = mt * 16 + r + 8 * hi;   // pixel row M of D element r
      float cv = c[r];
      float phi = 0.f;
      #pragma unroll
      for (int j = 0; j < NBAS; ++j) {
        float d = cv - s_mu[j];
        phi = fmaf(s_w[j], __expf(-50.f * d * d), phi);  // 1/(2*0.1^2) = 50
      }
      if (n < NFILT) s_phi[n * NPIX + pr] = s_usig[pr] * phi;  // invM folded in s_w
    }
  }
  __syncthreads();

  // conv2: NF->1 with flipped kernels, + reaction + clip
  int ty = tid >> 4, tx = tid & 15;
  int gy = oy + ty, gx = ox + tx;
  if (gy < HIMG && gx < WIMG) {
    float diff = 0.f;
    for (int n = 0; n < NFILT; ++n) {
      const float* sp = &s_phi[n * NPIX];
      const float* fl = &s_filt[n * 25];
      #pragma unroll
      for (int a2 = 0; a2 < 5; ++a2)
        #pragma unroll
        for (int b2 = 0; b2 < 5; ++b2)
          diff = fmaf(sp[(ty + a2) * HP + tx + b2], fl[24 - (a2 * 5 + b2)], diff);
    }
    float uu = s_u[(ty + 4) * USTR + tx + 4];
    float ff = f[b * HIMG * WIMG + gy * WIMG + gx];
    float lam = lamp[0];
    float reac = lam * (uu - ff) / (uu * uu + 1e-3f);
    float o = uu - diff - reac;
    o = fminf(fmaxf(o, 0.f), 1.f);
    out[b * HIMG * WIMG + gy * WIMG + gx] = o;
  }
}

extern "C" void kernel_launch(void* const* d_in, const int* in_sizes, int n_in,
                              void* d_out, int out_size, void* d_ws, size_t ws_size,
                              hipStream_t stream) {
  const float* u    = (const float*)d_in[0];
  const float* f    = (const float*)d_in[1];
  const float* filt = (const float*)d_in[2];
  const float* lam  = (const float*)d_in[3];
  const float* mu   = (const float*)d_in[4];
  const float* wts  = (const float*)d_in[5];
  float* out = (float*)d_out;

  float* w_usig = (float*)d_ws;           // TOTAL floats
  float* w_part = w_usig + TOTAL;         // block partials
  float* w_invM = w_part + 1024;          // 1 float

  int nb1 = (TOTAL + 255) / 256;          // 1013
  k_avgpool<<<nb1, 256, 0, stream>>>(u, w_usig, w_part);
  k_reduce<<<1, 256, 0, stream>>>(w_part, nb1, w_invM);
  int nb3 = BATCH * TILESX * TILESX;      // 1152 tiles
  k_fused<<<nb3, 256, 0, stream>>>(u, f, filt, lam, mu, wts, w_usig, w_invM, out);
}
